// STAR_73856257622161
// MI455X (gfx1250) — matrix operand order repaired and checked by reading, not compiled, not run on hardware
//
#include <hip/hip_runtime.h>
#include <hip/hip_bf16.h>
#include <stdint.h>

// ---------------------------------------------------------------------------
// STAR model forward on MI455X (gfx1250), wave32 + WMMA f16 + async-LDS path.
//   1) embed_pool : gather 13 sparse embeddings + mean-pool 50 seq -> Xh (B,896) f16
//   2) weight prep: f32 -> f16 TRANSPOSED (NxK) copies; STAR eff. weight fused
//   3) aux MLP  : 896->512->256->128 WMMA gemms (relu), 128->1 wave dot -> sa
//   4) STAR     : 4 masked WMMA gemms (B,896)x(896,1024), rows where sid==e
//   5) MLP      : 1024->512->256->128 WMMA gemms (relu), 128->1 wave dot -> mo
//   6) combine  : ctr = sigmoid(sa+mo) duplicated into d_out
//
// GEMM: block = 256 thr = 8 waves, tile 16(M) x 128(N), K-step 64, LDS holds
// FRAGMENT-READY operands (all accesses b64/b128). Tiles are staged with
// GLOBAL_LOAD_ASYNC_TO_LDS (ASYNCcnt) and double-buffered so the async DMA
// overlaps the WMMA math; falls back to load+ds_store if builtins absent.
// ---------------------------------------------------------------------------

typedef __attribute__((ext_vector_type(16))) _Float16 v16h;
typedef __attribute__((ext_vector_type(8)))  float    v8f;

#define BATCH   16384
#define FFEATS  13
#define VOCAB   10000
#define EMB     64
#define SEQLEN  50
#define INDIM   896     // 13*64 + 64
#define STARD   1024
#define NSLOTS  4

#if __has_builtin(__builtin_amdgcn_global_load_async_to_lds_b128) && \
    __has_builtin(__builtin_amdgcn_global_load_async_to_lds_b64)
#define USE_ASYNC_LDS 1
#else
#define USE_ASYNC_LDS 0
#endif

#if __has_builtin(__builtin_amdgcn_s_wait_asynccnt)
#define WAIT_ASYNC(n) __builtin_amdgcn_s_wait_asynccnt(n)
#else
#define WAIT_ASYNC(n) asm volatile("s_wait_asynccnt %0" ::"n"(n) : "memory")
#endif

#if USE_ASYNC_LDS
// Builtin signatures (from hipcc diagnostics):
//   b64 : (v2i AS(1)*, v2i AS(3)*, Ii offset, Ii cpol)
//   b128: (v4i AS(1)*, v4i AS(3)*, Ii offset, Ii cpol)
typedef int v2i __attribute__((vector_size(8)));
typedef int v4i __attribute__((vector_size(16)));
#define GB64(p)  ((__attribute__((address_space(1))) v2i*)(p))
#define LB64(p)  ((__attribute__((address_space(3))) v2i*)(p))
#define GB128(p) ((__attribute__((address_space(1))) v4i*)(p))
#define LB128(p) ((__attribute__((address_space(3))) v4i*)(p))
#endif

// ---------------------------------------------------------------------------
// 1) Embedding gather + sequence mean-pool -> f16 activations
// ---------------------------------------------------------------------------
__global__ void embed_pool_kernel(const int* __restrict__ sp_ids,
                                  const int* __restrict__ seq_ids,
                                  const float* __restrict__ sp_tables,
                                  const float* __restrict__ seq_table,
                                  _Float16* __restrict__ Xh) {
  const int b = blockIdx.x;
  const int t = threadIdx.x;          // 256 threads
  __shared__ int s_seq[SEQLEN];
  __shared__ int s_sp[FFEATS];
  if (t < SEQLEN) s_seq[t] = seq_ids[b * SEQLEN + t];
  if (t < FFEATS) s_sp[t]  = sp_ids[b * FFEATS + t];
  __syncthreads();
  for (int idx = t; idx < FFEATS * EMB; idx += 256) {
    const int f = idx >> 6;
    const int j = idx & 63;
    const float v = sp_tables[((size_t)f * VOCAB + s_sp[f]) * EMB + j];
    Xh[(size_t)b * INDIM + idx] = (_Float16)v;
  }
  if (t < EMB) {
    float acc = 0.f;
    for (int l = 0; l < SEQLEN; ++l)
      acc += seq_table[(size_t)s_seq[l] * EMB + t];
    Xh[(size_t)b * INDIM + FFEATS * EMB + t] = (_Float16)(acc * (1.0f / SEQLEN));
  }
}

// ---------------------------------------------------------------------------
// 2) Weight preparation: transpose KxN f32 -> NxK f16
// ---------------------------------------------------------------------------
__global__ void transpose_w_kernel(const float* __restrict__ src,  // K x N
                                   _Float16* __restrict__ dst,     // N x K
                                   int K, int N) {
  const int i = blockIdx.x * 256 + threadIdx.x;
  if (i < K * N) {
    const int n = i / K;
    const int k = i - n * K;
    dst[i] = (_Float16)src[(size_t)k * N + n];
  }
}

// WeffT[e][n][k] = slot_ws[e][k][n] * shared_w[k][n]   (f16, transposed)
__global__ void star_weight_t_kernel(const float* __restrict__ slot_ws,
                                     const float* __restrict__ shared_w,
                                     _Float16* __restrict__ wt) {
  const int i = blockIdx.x * 256 + threadIdx.x;
  const int per = INDIM * STARD;
  if (i < NSLOTS * per) {
    const int e = i / per;
    const int r = i - e * per;
    const int n = r / INDIM;          // 0..1023
    const int k = r - n * INDIM;      // 0..895
    const size_t kn = (size_t)k * STARD + n;
    wt[i] = (_Float16)(slot_ws[(size_t)e * per + kn] * shared_w[kn]);
  }
}

// biasE[e,n] = slot_bs[e,n] + shared_b[n]
__global__ void star_bias_kernel(const float* __restrict__ slot_bs,
                                 const float* __restrict__ shared_b,
                                 float* __restrict__ biasE) {
  const int i = blockIdx.x * 256 + threadIdx.x;
  if (i < NSLOTS * STARD) biasE[i] = slot_bs[i] + shared_b[i & (STARD - 1)];
}

// ---------------------------------------------------------------------------
// 3) WMMA GEMM:  Ch[M,N] = act(A[M,K] @ Wt[N,K]^T + bias)
//    A row-major f16, Wt TRANSPOSED (NxK) f16.
//    sid == nullptr -> store all rows; else store rows with sid[m]==slot.
// ---------------------------------------------------------------------------
__global__ __launch_bounds__(256)
void wmma_gemm_kernel(const _Float16* __restrict__ A,
                      const _Float16* __restrict__ Wt,
                      const float* __restrict__ bias,
                      const int* __restrict__ sid, int slot,
                      _Float16* __restrict__ Ch,
                      int M, int N, int K, int relu) {
  // fragment-ready LDS, double-buffered: [buf][kb][...]
  __shared__ __align__(32) _Float16 Afrag[2][2][16 * 32];       //  4 KB
  __shared__ __align__(32) _Float16 Bfrag[2][2][8][32 * 16];    // 32 KB
  const int t    = threadIdx.x;
  const int wave = t >> 5;
  const int lane = t & 31;
  const int m0   = blockIdx.y * 16;
  const int n0   = blockIdx.x * 128;
  const int g    = lane >> 4;
  const int l16  = lane & 15;

  // A staging: thread t owns 4 consecutive K-halfs of the 16x64 tile.
  // k' bits: e = k'[3:0], lane = m + 16*k'[4]  (contiguous A-operand map)
  const int a_idx  = t * 4;
  const int a_m    = a_idx >> 6;                 // 0..15
  const int a_k    = a_idx & 63;                 // 0..60 step 4
  const int a_kb   = a_k >> 5;
  const int a_kk   = a_k & 31;
  const int a_lane = a_m + 16 * (a_kk >> 4);
  const int a_e    = a_kk & 15;
  // B staging: thread t owns 32 K-halfs of weight row n (Wt is NxK).
  const int b_n  = t >> 1;                       // 0..127
  const int b_p  = t & 1;                        // which 32-K block
  const int b_w  = b_n >> 4;
  const int b_l0 = b_n & 15;

  const int nsteps = K >> 6;
  v8f c = {};

#if USE_ASYNC_LDS
  // ---- async DMA staging (GLOBAL_LOAD_ASYNC_TO_LDS, ASYNCcnt), pipelined ----
  auto issue_tile = [&](int buf, int k0) {
    // A: one b64 transfer into fragment-ready slot
    __builtin_amdgcn_global_load_async_to_lds_b64(
        GB64(&A[(size_t)(m0 + a_m) * K + k0 + a_k]),
        LB64(&Afrag[buf][a_kb][a_lane * 16 + a_e]), 0, 0);
    // B: 4 b128 transfers, 2 base pairs with matching src/dst imm offsets
    const _Float16* s = &Wt[(size_t)(n0 + b_n) * K + k0 + 32 * b_p];
    _Float16* d0 = &Bfrag[buf][b_p][b_w][b_l0 * 16];
    _Float16* d1 = &Bfrag[buf][b_p][b_w][(b_l0 + 16) * 16];
    __builtin_amdgcn_global_load_async_to_lds_b128(GB128(s),      LB128(d0), 0,  0);
    __builtin_amdgcn_global_load_async_to_lds_b128(GB128(s),      LB128(d0), 16, 0);
    __builtin_amdgcn_global_load_async_to_lds_b128(GB128(s + 16), LB128(d1), 0,  0);
    __builtin_amdgcn_global_load_async_to_lds_b128(GB128(s + 16), LB128(d1), 16, 0);
  };
  issue_tile(0, 0);
  for (int s = 0; s < nsteps; ++s) {
    const int buf = s & 1;
    if (s + 1 < nsteps) {
      issue_tile(buf ^ 1, (s + 1) << 6);   // overlap next tile's DMA with math
      WAIT_ASYNC(5);                       // in-order: current tile's 5 done
    } else {
      WAIT_ASYNC(0);
    }
    __syncthreads();
#pragma unroll
    for (int kb = 0; kb < 2; ++kb) {
      const v16h a = *(const v16h*)&Afrag[buf][kb][lane * 16];
      const v16h b = *(const v16h*)&Bfrag[buf][kb][wave][lane * 16];
      c = __builtin_amdgcn_wmma_f32_16x16x32_f16(
          false, a, false, b, (short)0, c, false, false);
    }
    __syncthreads();   // readers done before buf is re-staged
  }
#else
  // ---- fallback: coalesced load + ds_store staging, single buffer ----
  for (int s = 0; s < nsteps; ++s) {
    const int k0 = s << 6;
    *(unsigned long long*)&Afrag[0][a_kb][a_lane * 16 + a_e] =
        *(const unsigned long long*)&A[(size_t)(m0 + a_m) * K + k0 + a_k];
    {
      const uint4* src = (const uint4*)&Wt[(size_t)(n0 + b_n) * K + k0 + 32 * b_p];
      uint4* d0 = (uint4*)&Bfrag[0][b_p][b_w][b_l0 * 16];
      d0[0] = src[0];
      d0[1] = src[1];
      uint4* d1 = (uint4*)&Bfrag[0][b_p][b_w][(b_l0 + 16) * 16];
      d1[0] = src[2];
      d1[1] = src[3];
    }
    if (k0 + 64 < K) {
      __builtin_prefetch(&Wt[(size_t)(n0 + b_n) * K + k0 + 64 + 32 * b_p], 0, 1);
    }
    __syncthreads();
#pragma unroll
    for (int kb = 0; kb < 2; ++kb) {
      const v16h a = *(const v16h*)&Afrag[0][kb][lane * 16];
      const v16h b = *(const v16h*)&Bfrag[0][kb][wave][lane * 16];
      c = __builtin_amdgcn_wmma_f32_16x16x32_f16(
          false, a, false, b, (short)0, c, false, false);
    }
    __syncthreads();
  }
#endif

  // D layout: VGPR r -> row m0 + r + 8*g ; col n0 + wave*16 + (lane&15)
  const int n = n0 + wave * 16 + l16;
  const float bv = bias ? bias[n] : 0.0f;
#pragma unroll
  for (int r = 0; r < 8; ++r) {
    const int m = m0 + r + 8 * g;
    float v = c[r] + bv;
    if (relu) v = v > 0.f ? v : 0.f;
    if (!sid || sid[m] == slot) Ch[(size_t)m * N + n] = (_Float16)v;
  }
}

// ---------------------------------------------------------------------------
// 4) Final 128 -> 1 layer: one wave per row, shfl reduction (wave32)
// ---------------------------------------------------------------------------
__global__ void final_dot_kernel(const _Float16* __restrict__ H,
                                 const float* __restrict__ Wl,
                                 const float* __restrict__ bl,
                                 float* __restrict__ out, int M) {
  const int row  = blockIdx.x * 8 + (threadIdx.x >> 5);
  const int lane = threadIdx.x & 31;
  if (row >= M) return;
  float acc = 0.f;
#pragma unroll
  for (int i = lane; i < 128; i += 32)
    acc += (float)H[(size_t)row * 128 + i] * Wl[i];
#pragma unroll
  for (int off = 16; off; off >>= 1)
    acc += __shfl_xor(acc, off, 32);
  if (lane == 0) out[row] = acc + bl[0];
}

// ---------------------------------------------------------------------------
// 5) ctr = sigmoid(sa + mo), duplicated tuple output
// ---------------------------------------------------------------------------
__global__ void combine_kernel(const float* __restrict__ sa,
                               const float* __restrict__ mo,
                               float* __restrict__ out, int Bn) {
  const int i = blockIdx.x * 256 + threadIdx.x;
  if (i < Bn) {
    const float v = 1.0f / (1.0f + expf(-(sa[i] + mo[i])));
    out[i] = v;
    out[Bn + i] = v;
  }
}

// ---------------------------------------------------------------------------
// Host launcher
// ---------------------------------------------------------------------------
extern "C" void kernel_launch(void* const* d_in, const int* in_sizes, int n_in,
                              void* d_out, int out_size, void* d_ws, size_t ws_size,
                              hipStream_t stream) {
  const int*   sparse_inputs   = (const int*)d_in[0];
  const int*   sequence_inputs = (const int*)d_in[1];
  const int*   scenario_id     = (const int*)d_in[2];
  const float* sparse_tables   = (const float*)d_in[3];
  const float* seq_table       = (const float*)d_in[4];
  const float* shared_w        = (const float*)d_in[5];
  const float* shared_b        = (const float*)d_in[6];
  const float* slot_ws         = (const float*)d_in[7];
  const float* slot_bs         = (const float*)d_in[8];
  const float* auxW[4] = {(const float*)d_in[9],  (const float*)d_in[10],
                          (const float*)d_in[11], (const float*)d_in[12]};
  const float* auxB[4] = {(const float*)d_in[13], (const float*)d_in[14],
                          (const float*)d_in[15], (const float*)d_in[16]};
  const float* mlpW[4] = {(const float*)d_in[17], (const float*)d_in[18],
                          (const float*)d_in[19], (const float*)d_in[20]};
  const float* mlpB[4] = {(const float*)d_in[21], (const float*)d_in[22],
                          (const float*)d_in[23], (const float*)d_in[24]};
  float* out = (float*)d_out;

  // --- workspace carve-out (256B aligned) ---
  char* base = (char*)d_ws;
  size_t off = 0;
  auto carve = [&](size_t bytes) -> char* {
    char* p = base + off;
    off = (off + bytes + 255) & ~(size_t)255;
    return p;
  };
  _Float16* Xh     = (_Float16*)carve((size_t)BATCH * INDIM * 2);
  _Float16* aw0    = (_Float16*)carve((size_t)896 * 512 * 2);    // transposed NxK
  _Float16* aw1    = (_Float16*)carve((size_t)512 * 256 * 2);
  _Float16* aw2    = (_Float16*)carve((size_t)256 * 128 * 2);
  _Float16* mw0    = (_Float16*)carve((size_t)1024 * 512 * 2);
  _Float16* mw1    = (_Float16*)carve((size_t)512 * 256 * 2);
  _Float16* mw2    = (_Float16*)carve((size_t)256 * 128 * 2);
  _Float16* weff   = (_Float16*)carve((size_t)NSLOTS * INDIM * STARD * 2);
  float*    biasE  = (float*)   carve((size_t)NSLOTS * STARD * 4);
  _Float16* h512   = (_Float16*)carve((size_t)BATCH * 512 * 2);  // reused aux/mlp
  _Float16* h256   = (_Float16*)carve((size_t)BATCH * 256 * 2);
  _Float16* h128   = (_Float16*)carve((size_t)BATCH * 128 * 2);
  float*    sa     = (float*)   carve((size_t)BATCH * 4);
  _Float16* star_h = (_Float16*)carve((size_t)BATCH * STARD * 2);
  float*    mo     = (float*)   carve((size_t)BATCH * 4);
  (void)ws_size; (void)n_in; (void)in_sizes; (void)out_size;

  // 1) embeddings -> Xh
  embed_pool_kernel<<<BATCH, 256, 0, stream>>>(
      sparse_inputs, sequence_inputs, sparse_tables, seq_table, Xh);

  // 2) weight prep (transposed f16 copies)
  auto cvtT = [&](const float* s, _Float16* d, int K, int N) {
    transpose_w_kernel<<<(K * N + 255) / 256, 256, 0, stream>>>(s, d, K, N);
  };
  cvtT(auxW[0], aw0, 896, 512);
  cvtT(auxW[1], aw1, 512, 256);
  cvtT(auxW[2], aw2, 256, 128);
  cvtT(mlpW[0], mw0, 1024, 512);
  cvtT(mlpW[1], mw1, 512, 256);
  cvtT(mlpW[2], mw2, 256, 128);
  star_weight_t_kernel<<<(NSLOTS * INDIM * STARD + 255) / 256, 256, 0, stream>>>(
      slot_ws, shared_w, weff);
  star_bias_kernel<<<(NSLOTS * STARD + 255) / 256, 256, 0, stream>>>(
      slot_bs, shared_b, biasE);

  const int MT = BATCH / 16;  // 1024 M-tiles

  // 3) aux tower: 896->512->256->128 (relu), 128->1
  wmma_gemm_kernel<<<dim3(512 / 128, MT), 256, 0, stream>>>(
      Xh, aw0, auxB[0], nullptr, 0, h512, BATCH, 512, 896, 1);
  wmma_gemm_kernel<<<dim3(256 / 128, MT), 256, 0, stream>>>(
      h512, aw1, auxB[1], nullptr, 0, h256, BATCH, 256, 512, 1);
  wmma_gemm_kernel<<<dim3(128 / 128, MT), 256, 0, stream>>>(
      h256, aw2, auxB[2], nullptr, 0, h128, BATCH, 128, 256, 1);
  final_dot_kernel<<<BATCH / 8, 256, 0, stream>>>(h128, auxW[3], auxB[3], sa, BATCH);

  // 4) STAR: 4 masked GEMMs; each slot's 1.8MB f16 weight stays L2-resident
  for (int e = 0; e < NSLOTS; ++e) {
    wmma_gemm_kernel<<<dim3(STARD / 128, MT), 256, 0, stream>>>(
        Xh, weff + (size_t)e * INDIM * STARD, biasE + e * STARD,
        scenario_id, e, star_h, BATCH, STARD, INDIM, 0);
  }

  // 5) main tower: 1024->512->256->128 (relu), 128->1
  wmma_gemm_kernel<<<dim3(512 / 128, MT), 256, 0, stream>>>(
      star_h, mw0, mlpB[0], nullptr, 0, h512, BATCH, 512, 1024, 1);
  wmma_gemm_kernel<<<dim3(256 / 128, MT), 256, 0, stream>>>(
      h512, mw1, mlpB[1], nullptr, 0, h256, BATCH, 256, 512, 1);
  wmma_gemm_kernel<<<dim3(128 / 128, MT), 256, 0, stream>>>(
      h256, mw2, mlpB[2], nullptr, 0, h128, BATCH, 128, 256, 1);
  final_dot_kernel<<<BATCH / 8, 256, 0, stream>>>(h128, mlpW[3], mlpB[3], mo, BATCH);

  // 6) combine
  combine_kernel<<<(BATCH + 255) / 256, 256, 0, stream>>>(sa, mo, out, BATCH);
}
